// DecoderGRU_18966575579787
// MI455X (gfx1250) — matrix operand order repaired and checked by reading, not compiled, not run on hardware
//
#include <hip/hip_runtime.h>
#include <math.h>

#define E_DIM 512
#define H_DIM 512
#define V_DIM 10000
#define B_DIM 64
#define R_DIM 49
#define L_DIM 33
#define T_DIM 32

typedef __bf16 bf16;
typedef bf16  v16bf __attribute__((ext_vector_type(16)));
typedef bf16  v8bf  __attribute__((ext_vector_type(8)));
typedef float v8f   __attribute__((ext_vector_type(8)));

// ---------------------------------------------------------------------------
// V_WMMA_F32_16X16X32_BF16 fragment helpers (wave32), ISA 7.12.2:
//  A 16x32 (MxK), lane l: m = l&15, half = l>>4.
//    this code puts K = half*16 + {0..15} in VGPR 0..7 (contiguous, not ISA table order)
//    -> one contiguous 32-byte run per lane from a row-major (M x K) source.
//  B 32x16 (KxN), lane l: n = l&15, half = l>>4.
//    here: VGPR j holds K = half*16 + {2j, 2j+1} (as A, not ISA order) -> one contiguous 32-byte run of K
//    per lane from an N-major (N x K) source.
//  C/D 16x16: lane = n + 16*(m>=8), vgpr = m&7.
// ---------------------------------------------------------------------------
__device__ __forceinline__ v16bf ld_a_bf(const bf16* __restrict__ A, int lda,
                                         int row0, int k0, int lane) {
  int m = lane & 15, half = lane >> 4;
  const bf16* p = A + (size_t)(row0 + m) * lda + k0 + half * 16;
  union { v16bf v; v8bf h[2]; } u;
  u.h[0] = *(const v8bf*)p;          // K = half*16 + 0..7
  u.h[1] = *(const v8bf*)(p + 8);    // K = half*16 + 8..15
  return u.v;
}

// Wt is (N x K) row-major ("transposed weight"), leading dim ldk
__device__ __forceinline__ v16bf ld_bt_bf(const bf16* __restrict__ Wt, int ldk,
                                          int col0, int k0, int lane) {
  int n = lane & 15, half = lane >> 4;
  const bf16* p = Wt + (size_t)(col0 + n) * ldk + k0 + half * 16;
  return *(const v16bf*)p;           // K = half*16 + 0..15 ascending
}

#define WMMA_BF16(acc, a, b)                                                   \
  (acc) = __builtin_amdgcn_wmma_f32_16x16x32_bf16(false, (a), false, (b),      \
                                                  (short)0, (acc), false, false)

// split-bf16 product: acc += (ah+al) x (bh+bl), dropping the lo*lo term
#define WMMA_SPLIT3(acc, ah, al, bh, bl)                                       \
  do {                                                                         \
    WMMA_BF16(acc, ah, bh);                                                    \
    WMMA_BF16(acc, ah, bl);                                                    \
    WMMA_BF16(acc, al, bh);                                                    \
  } while (0)

// Async global->LDS copy, 16 bytes per lane (ASYNCcnt-tracked TDM-lite path).
// vdst = per-lane LDS byte offset, vaddr = per-lane 64-bit global address.
__device__ __forceinline__ void async_copy_b128(unsigned lds_off, const void* gptr) {
  asm volatile("global_load_async_to_lds_b128 %0, %1, off"
               :: "v"(lds_off), "v"(gptr) : "memory");
}
__device__ __forceinline__ void wait_async0() {
  asm volatile("s_wait_asynccnt 0x0" ::: "memory");
}

__device__ __forceinline__ float sigmoidf_(float x) {
  return 1.0f / (1.0f + expf(-x));
}

__device__ __forceinline__ void split_f32(float x, bf16& hi, bf16& lo) {
  bf16 h = (bf16)x;
  hi = h;
  lo = (bf16)(x - (float)h);
}

// ---------------------------------------------------------------------------
// Conversion / utility kernels
// ---------------------------------------------------------------------------
__global__ void zero_bf16_kernel(bf16* __restrict__ p, int n) {
  int i = blockIdx.x * blockDim.x + threadIdx.x;
  if (i < n) p[i] = (bf16)0.0f;
}

// plain elementwise hi/lo split (layout preserved)
__global__ void split_kernel(const float* __restrict__ src,
                             bf16* __restrict__ hi, bf16* __restrict__ lo, int n) {
  int i = blockIdx.x * blockDim.x + threadIdx.x;
  if (i >= n) return;
  split_f32(src[i], hi[i], lo[i]);
}

// attn_W (2E x H, K-major) -> two transposed (H x E) hi/lo pairs
__global__ void split_attnW_kernel(const float* __restrict__ attn_W,
                                   bf16* __restrict__ w1h, bf16* __restrict__ w1l,
                                   bf16* __restrict__ w2h, bf16* __restrict__ w2l) {
  int i = blockIdx.x * blockDim.x + threadIdx.x;
  if (i >= E_DIM * H_DIM) return;
  int n = i % H_DIM;           // coalesced read along n
  int k = i / H_DIM;
  size_t o = (size_t)n * E_DIM + k;
  split_f32(attn_W[(size_t)k * H_DIM + n], w1h[o], w1l[o]);
  split_f32(attn_W[(size_t)(E_DIM + k) * H_DIM + n], w2h[o], w2l[o]);
}

// fc_W (H x V) -> transposed (V x H) hi/lo
__global__ void split_fcW_kernel(const float* __restrict__ fc_W,
                                 bf16* __restrict__ wth, bf16* __restrict__ wtl) {
  int i = blockIdx.x * blockDim.x + threadIdx.x;
  if (i >= H_DIM * V_DIM) return;
  int n = i % V_DIM;           // coalesced read along n
  int k = i / V_DIM;
  size_t o = (size_t)n * H_DIM + k;
  split_f32(fc_W[(size_t)k * V_DIM + n], wth[o], wtl[o]);
}

// X[t][b][e] = split(embed_table[captions[b*L + t]][e]),  t in [0,T)
__global__ void gather_split_kernel(const int* __restrict__ captions,
                                    const float* __restrict__ table,
                                    bf16* __restrict__ Xh, bf16* __restrict__ Xl) {
  int idx = blockIdx.x * blockDim.x + threadIdx.x;
  if (idx >= T_DIM * B_DIM * E_DIM) return;
  int e  = idx % E_DIM;
  int tb = idx / E_DIM;
  int b  = tb % B_DIM;
  int t  = tb / B_DIM;
  int token = captions[b * L_DIM + t];
  split_f32(table[(size_t)token * E_DIM + e], Xh[idx], Xl[idx]);
}

// ---------------------------------------------------------------------------
// feat_proj(f32) = features(3136x512) @ attn_W[0:E,:] + attn_b   [bf16 split]
// One 16x16 tile/wave; 196*32 = 6272 tiles; 8 waves/block -> 784 blocks.
// ---------------------------------------------------------------------------
__global__ void featproj_kernel(const bf16* __restrict__ fh, const bf16* __restrict__ fl,
                                const bf16* __restrict__ w1h, const bf16* __restrict__ w1l,
                                const float* __restrict__ attn_b,
                                float* __restrict__ feat_proj) {
  const int lane = threadIdx.x & 31;
  const int wave = threadIdx.x >> 5;
  int tile = blockIdx.x * 8 + wave;
  const int NT = H_DIM / 16;
  int tm = tile / NT, tn = tile % NT;
  int row0 = tm * 16, col0 = tn * 16;
  v8f acc = {};
  for (int k0 = 0; k0 < E_DIM; k0 += 32) {
    v16bf ah = ld_a_bf(fh, E_DIM, row0, k0, lane);
    v16bf al = ld_a_bf(fl, E_DIM, row0, k0, lane);
    v16bf bh = ld_bt_bf(w1h, E_DIM, col0, k0, lane);
    v16bf bl = ld_bt_bf(w1l, E_DIM, col0, k0, lane);
    WMMA_SPLIT3(acc, ah, al, bh, bl);
  }
  int n  = lane & 15;
  int mh = (lane >> 4) << 3;
  float bias = attn_b[col0 + n];
#pragma unroll
  for (int v = 0; v < 8; ++v)
    feat_proj[(size_t)(row0 + mh + v) * H_DIM + col0 + n] = acc[v] + bias;
}

// ---------------------------------------------------------------------------
// hproj(f32) = h(64x512) @ attn_W[E:,:]   [bf16 split]; 128 tiles -> 16 blocks
// ---------------------------------------------------------------------------
__global__ void hproj_kernel(const bf16* __restrict__ hh, const bf16* __restrict__ hl,
                             const bf16* __restrict__ w2h, const bf16* __restrict__ w2l,
                             float* __restrict__ hproj) {
  const int lane = threadIdx.x & 31;
  const int wave = threadIdx.x >> 5;
  int tile = blockIdx.x * 8 + wave;
  const int NT = H_DIM / 16;
  int tm = tile / NT, tn = tile % NT;
  int row0 = tm * 16, col0 = tn * 16;
  v8f acc = {};
  for (int k0 = 0; k0 < H_DIM; k0 += 32) {
    v16bf ah = ld_a_bf(hh, H_DIM, row0, k0, lane);
    v16bf al = ld_a_bf(hl, H_DIM, row0, k0, lane);
    v16bf bh = ld_bt_bf(w2h, H_DIM, col0, k0, lane);
    v16bf bl = ld_bt_bf(w2l, H_DIM, col0, k0, lane);
    WMMA_SPLIT3(acc, ah, al, bh, bl);
  }
  int n  = lane & 15;
  int mh = (lane >> 4) << 3;
#pragma unroll
  for (int v = 0; v < 8; ++v)
    hproj[(size_t)(row0 + mh + v) * H_DIM + col0 + n] = acc[v];
}

// ---------------------------------------------------------------------------
// Per-batch attention (f32 VALU): scores = tanh(feat_proj+hproj).v_w,
// softmax over R, context = attn @ features, written as bf16 hi/lo split.
// One workgroup (8 waves) per b.
// ---------------------------------------------------------------------------
__global__ void attn_kernel(const float* __restrict__ feat_proj,
                            const float* __restrict__ hproj,
                            const float* __restrict__ v_w,
                            const float* __restrict__ features,
                            bf16* __restrict__ ch, bf16* __restrict__ cl) {
  __shared__ float s_sc[R_DIM];
  __shared__ float s_inv;
  int b = blockIdx.x;
  int tid = threadIdx.x;
  int lane = tid & 31, wave = tid >> 5;

  const float* hp = hproj + (size_t)b * H_DIM;
  for (int r = wave; r < R_DIM; r += 8) {
    const float* fp = feat_proj + (size_t)(b * R_DIM + r) * H_DIM;
    float p = 0.0f;
    for (int h = lane; h < H_DIM; h += 32)
      p += tanhf(fp[h] + hp[h]) * v_w[h];
    for (int off = 16; off > 0; off >>= 1)
      p += __shfl_down(p, off, 32);
    if (lane == 0) s_sc[r] = p;
  }
  __syncthreads();

  if (tid == 0) {
    float mx = s_sc[0];
    for (int r = 1; r < R_DIM; ++r) mx = fmaxf(mx, s_sc[r]);
    float sum = 0.0f;
    for (int r = 0; r < R_DIM; ++r) {
      float e = expf(s_sc[r] - mx);
      s_sc[r] = e;
      sum += e;
    }
    s_inv = 1.0f / sum;
  }
  __syncthreads();

  float inv = s_inv;
  for (int e = tid; e < E_DIM; e += blockDim.x) {
    float acc = 0.0f;
    for (int r = 0; r < R_DIM; ++r)
      acc += s_sc[r] * features[(size_t)(b * R_DIM + r) * E_DIM + e];
    size_t o = (size_t)b * E_DIM + e;
    split_f32(acc * inv, ch[o], cl[o]);
  }
}

// ---------------------------------------------------------------------------
// Fused GRU gates [bf16 split]: 6 f32 accumulators per wave —
// gx{r,z,n} over K=1024 with A=[x_t | ctx], gh{r,z,n} over K=512 with A=h.
// Epilogue reconstructs h_prev = hi+lo, applies gates, writes h_new hi/lo.
// 128 tiles -> 16 blocks of 8 waves.
// ---------------------------------------------------------------------------
__global__ void gru_gates_kernel(const bf16* __restrict__ Xth, const bf16* __restrict__ Xtl,
                                 const bf16* __restrict__ ch,  const bf16* __restrict__ cl,
                                 const bf16* __restrict__ hph, const bf16* __restrict__ hpl,
                                 const bf16* __restrict__ wihh, const bf16* __restrict__ wihl,
                                 const bf16* __restrict__ whhh, const bf16* __restrict__ whhl,
                                 const float* __restrict__ b_ih, const float* __restrict__ b_hh,
                                 bf16* __restrict__ hoh, bf16* __restrict__ hol) {
  const int lane = threadIdx.x & 31;
  const int wave = threadIdx.x >> 5;
  int tile = blockIdx.x * 8 + wave;
  const int NT = H_DIM / 16;
  int tm = tile / NT, tn = tile % NT;
  int row0 = tm * 16, col0 = tn * 16;

  v8f axr = {}, axz = {}, axn = {}, ahr = {}, ahz = {}, ahn = {};

  // gx = [x_t | ctx] @ W_ih^T  (W_ih already N-major: (3H x 2E))
  for (int k0 = 0; k0 < 2 * E_DIM; k0 += 32) {
    const bf16* Ah = (k0 < E_DIM) ? Xth : ch;
    const bf16* Al = (k0 < E_DIM) ? Xtl : cl;
    int kk = (k0 < E_DIM) ? k0 : (k0 - E_DIM);
    v16bf ah = ld_a_bf(Ah, E_DIM, row0, kk, lane);
    v16bf al = ld_a_bf(Al, E_DIM, row0, kk, lane);
    {
      v16bf bh = ld_bt_bf(wihh, 2 * E_DIM, col0, k0, lane);
      v16bf bl = ld_bt_bf(wihl, 2 * E_DIM, col0, k0, lane);
      WMMA_SPLIT3(axr, ah, al, bh, bl);
    }
    {
      v16bf bh = ld_bt_bf(wihh, 2 * E_DIM, col0 + H_DIM, k0, lane);
      v16bf bl = ld_bt_bf(wihl, 2 * E_DIM, col0 + H_DIM, k0, lane);
      WMMA_SPLIT3(axz, ah, al, bh, bl);
    }
    {
      v16bf bh = ld_bt_bf(wihh, 2 * E_DIM, col0 + 2 * H_DIM, k0, lane);
      v16bf bl = ld_bt_bf(wihl, 2 * E_DIM, col0 + 2 * H_DIM, k0, lane);
      WMMA_SPLIT3(axn, ah, al, bh, bl);
    }
  }
  // gh = h_prev @ W_hh^T  (W_hh already N-major: (3H x H))
  for (int k0 = 0; k0 < H_DIM; k0 += 32) {
    v16bf ah = ld_a_bf(hph, H_DIM, row0, k0, lane);
    v16bf al = ld_a_bf(hpl, H_DIM, row0, k0, lane);
    {
      v16bf bh = ld_bt_bf(whhh, H_DIM, col0, k0, lane);
      v16bf bl = ld_bt_bf(whhl, H_DIM, col0, k0, lane);
      WMMA_SPLIT3(ahr, ah, al, bh, bl);
    }
    {
      v16bf bh = ld_bt_bf(whhh, H_DIM, col0 + H_DIM, k0, lane);
      v16bf bl = ld_bt_bf(whhl, H_DIM, col0 + H_DIM, k0, lane);
      WMMA_SPLIT3(ahz, ah, al, bh, bl);
    }
    {
      v16bf bh = ld_bt_bf(whhh, H_DIM, col0 + 2 * H_DIM, k0, lane);
      v16bf bl = ld_bt_bf(whhl, H_DIM, col0 + 2 * H_DIM, k0, lane);
      WMMA_SPLIT3(ahn, ah, al, bh, bl);
    }
  }

  int n  = lane & 15;
  int mh = (lane >> 4) << 3;
  int nc = col0 + n;
  float bir = b_ih[nc], biz = b_ih[nc + H_DIM], bin = b_ih[nc + 2 * H_DIM];
  float bhr = b_hh[nc], bhz = b_hh[nc + H_DIM], bhn = b_hh[nc + 2 * H_DIM];
#pragma unroll
  for (int v = 0; v < 8; ++v) {
    size_t o = (size_t)(row0 + mh + v) * H_DIM + nc;
    float r  = sigmoidf_((axr[v] + bir) + (ahr[v] + bhr));
    float z  = sigmoidf_((axz[v] + biz) + (ahz[v] + bhz));
    float nn = tanhf((axn[v] + bin) + r * (ahn[v] + bhn));
    float hp = (float)hph[o] + (float)hpl[o];
    split_f32((1.0f - z) * nn + z * hp, hoh[o], hol[o]);
  }
}

// ---------------------------------------------------------------------------
// Final classifier [bf16 split]: out = H_all(2048x512) @ fc_W + fc_b.
// Grid (125 col-groups, 16 row-groups). The 8 waves of a block share one
// 80-column B panel: it is staged per 32-K chunk into LDS with
// global_load_async_to_lds_b128 (ASYNCcnt) and consumed 8x via ds_load_b128.
// A fragments stay in registers (15x reuse per chunk).
// ---------------------------------------------------------------------------
__global__ void fc_kernel(const bf16* __restrict__ Hh, const bf16* __restrict__ Hl,
                          const bf16* __restrict__ wth, const bf16* __restrict__ wtl,
                          const float* __restrict__ fc_b,
                          float* __restrict__ out) {
  __shared__ __align__(64) bf16 sBh[80 * 32];   // [panel col][k] hi
  __shared__ __align__(64) bf16 sBl[80 * 32];   // [panel col][k] lo
  const int lane = threadIdx.x & 31;
  const int wave = threadIdx.x >> 5;
  const int col0 = blockIdx.x * 80;                   // 125 col groups
  const int row0 = (blockIdx.y * 8 + wave) * 16;      // 16 row groups x 8 waves
  const int n = lane & 15, half = lane >> 4;

  v8f acc[5] = {v8f{}, v8f{}, v8f{}, v8f{}, v8f{}};
  for (int k0 = 0; k0 < H_DIM; k0 += 32) {
    // cooperative async stage of the B panel: 640 x 16B transfers (hi + lo)
    for (int i = threadIdx.x; i < 640; i += 256) {
      int which = (i >= 320);
      int j = which ? (i - 320) : i;
      int col = j >> 2, seg = j & 3;              // 4 x 16B per 64B panel row
      const bf16* g = (which ? wtl : wth) +
                      (size_t)(col0 + col) * H_DIM + k0 + seg * 8;
      bf16* l = (which ? sBl : sBh) + col * 32 + seg * 8;
      async_copy_b128((unsigned)(size_t)l, (const void*)g);
    }
    wait_async0();
    __syncthreads();

    v16bf ah = ld_a_bf(Hh, H_DIM, row0, k0, lane);
    v16bf al = ld_a_bf(Hl, H_DIM, row0, k0, lane);
#pragma unroll
    for (int s = 0; s < 5; ++s) {
      v16bf bh = *(const v16bf*)(sBh + (s * 16 + n) * 32 + half * 16);
      v16bf bl = *(const v16bf*)(sBl + (s * 16 + n) * 32 + half * 16);
      WMMA_SPLIT3(acc[s], ah, al, bh, bl);
    }
    __syncthreads();   // single-buffered panel: protect against next overwrite
  }

  int mh = half << 3;
#pragma unroll
  for (int s = 0; s < 5; ++s) {
    int col = col0 + s * 16 + n;
    float bias = fc_b[col];
#pragma unroll
    for (int v = 0; v < 8; ++v) {
      int row = row0 + mh + v;
      int t = row >> 6;        // row = t*64 + b
      int b = row & 63;
      out[(size_t)b * (T_DIM * V_DIM) + (size_t)t * V_DIM + col] = acc[s][v] + bias;
    }
  }
}

// ---------------------------------------------------------------------------
// Host orchestration
// ---------------------------------------------------------------------------
static inline size_t rnd256(size_t x) { return (x + 255) & ~(size_t)255; }

extern "C" void kernel_launch(void* const* d_in, const int* in_sizes, int n_in,
                              void* d_out, int out_size, void* d_ws, size_t ws_size,
                              hipStream_t stream) {
  const float* features    = (const float*)d_in[0];
  const int*   captions    = (const int*)  d_in[1];
  const float* embed_table = (const float*)d_in[2];
  const float* attn_W      = (const float*)d_in[3];
  const float* attn_b      = (const float*)d_in[4];
  const float* v_w         = (const float*)d_in[5];
  const float* W_ih        = (const float*)d_in[6];
  const float* W_hh        = (const float*)d_in[7];
  const float* b_ih        = (const float*)d_in[8];
  const float* b_hh        = (const float*)d_in[9];
  const float* fc_W        = (const float*)d_in[10];
  const float* fc_b        = (const float*)d_in[11];
  float* out = (float*)d_out;

  char* ws = (char*)d_ws;
  size_t off = 0;
  auto alloc = [&](size_t bytes) { char* p = ws + off; off += rnd256(bytes); return p; };

  const size_t TBE = (size_t)T_DIM * B_DIM * E_DIM;     // 1,048,576
  const size_t BRH = (size_t)B_DIM * R_DIM * H_DIM;     // 1,605,632
  const size_t TBH = (size_t)T_DIM * B_DIM * H_DIM;     // 1,048,576
  const size_t BH  = (size_t)B_DIM * H_DIM;             // 32,768

  bf16* Xh   = (bf16*)alloc(TBE * 2);
  bf16* Xl   = (bf16*)alloc(TBE * 2);
  bf16* Fh   = (bf16*)alloc(BRH * 2);
  bf16* Fl   = (bf16*)alloc(BRH * 2);
  bf16* W1h  = (bf16*)alloc((size_t)E_DIM * H_DIM * 2);
  bf16* W1l  = (bf16*)alloc((size_t)E_DIM * H_DIM * 2);
  bf16* W2h  = (bf16*)alloc((size_t)E_DIM * H_DIM * 2);
  bf16* W2l  = (bf16*)alloc((size_t)E_DIM * H_DIM * 2);
  bf16* Wihh = (bf16*)alloc((size_t)3 * H_DIM * 2 * E_DIM * 2);
  bf16* Wihl = (bf16*)alloc((size_t)3 * H_DIM * 2 * E_DIM * 2);
  bf16* Whhh = (bf16*)alloc((size_t)3 * H_DIM * H_DIM * 2);
  bf16* Whhl = (bf16*)alloc((size_t)3 * H_DIM * H_DIM * 2);
  bf16* Wth  = (bf16*)alloc((size_t)V_DIM * H_DIM * 2);
  bf16* Wtl  = (bf16*)alloc((size_t)V_DIM * H_DIM * 2);
  bf16* Ch   = (bf16*)alloc(BH * 2);
  bf16* Cl   = (bf16*)alloc(BH * 2);
  bf16* Hh   = (bf16*)alloc(TBH * 2);
  bf16* Hl   = (bf16*)alloc(TBH * 2);
  bf16* Z0   = (bf16*)alloc(BH * 2);        // shared zero hi/lo for h_{-1}
  float* FP  = (float*)alloc(BRH * 4);      // feat_proj (f32)
  float* HP  = (float*)alloc(BH * 4);       // hproj (f32)

  auto blocks = [](size_t n) { return (unsigned)((n + 255) / 256); };

  zero_bf16_kernel<<<blocks(BH), 256, 0, stream>>>(Z0, (int)BH);
  gather_split_kernel<<<blocks(TBE), 256, 0, stream>>>(captions, embed_table, Xh, Xl);
  split_kernel<<<blocks(BRH), 256, 0, stream>>>(features, Fh, Fl, (int)BRH);
  split_attnW_kernel<<<blocks((size_t)E_DIM * H_DIM), 256, 0, stream>>>(
      attn_W, W1h, W1l, W2h, W2l);
  split_kernel<<<blocks((size_t)3 * H_DIM * 2 * E_DIM), 256, 0, stream>>>(
      W_ih, Wihh, Wihl, 3 * H_DIM * 2 * E_DIM);
  split_kernel<<<blocks((size_t)3 * H_DIM * H_DIM), 256, 0, stream>>>(
      W_hh, Whhh, Whhl, 3 * H_DIM * H_DIM);
  split_fcW_kernel<<<blocks((size_t)H_DIM * V_DIM), 256, 0, stream>>>(fc_W, Wth, Wtl);

  featproj_kernel<<<784, 256, 0, stream>>>(Fh, Fl, W1h, W1l, attn_b, FP);

  for (int t = 0; t < T_DIM; ++t) {
    const bf16* hph = (t == 0) ? Z0 : (Hh + (size_t)(t - 1) * BH);
    const bf16* hpl = (t == 0) ? Z0 : (Hl + (size_t)(t - 1) * BH);
    bf16* hoh = Hh + (size_t)t * BH;
    bf16* hol = Hl + (size_t)t * BH;
    hproj_kernel<<<16, 256, 0, stream>>>(hph, hpl, W2h, W2l, HP);
    attn_kernel<<<64, 256, 0, stream>>>(FP, HP, v_w, features, Ch, Cl);
    gru_gates_kernel<<<16, 256, 0, stream>>>(
        Xh + (size_t)t * B_DIM * E_DIM, Xl + (size_t)t * B_DIM * E_DIM,
        Ch, Cl, hph, hpl, Wihh, Wihl, Whhh, Whhl, b_ih, b_hh, hoh, hol);
  }

  dim3 fc_grid(125, 16);
  fc_kernel<<<fc_grid, 256, 0, stream>>>(Hh, Hl, Wth, Wtl, fc_b, out);
}
